// PocketGraphProcessor_5497558139215
// MI455X (gfx1250) — matrix operand order repaired and checked by reading, not compiled, not run on hardware
//
#include <hip/hip_runtime.h>
#include <hip/hip_bf16.h>
#include <math.h>

#define NN 50000
#define EE 400000
#define GG 256
#define DD 256
#define VDIM 16

// K dims (padded to multiples of 32 for wmma_f32_16x16x32_bf16)
#define K1 800   // 256 (s_dst) + 256 (s_src) + 16 (vnorm) + 16 (pad) + 256 (e)
#define K2 256
#define K3 288   // 256 (s) + 16 (vnorm) + 16 (pad)
#define KT1 (K1/32)
#define KT2 (K2/32)
#define KT3 (K3/32)

// Edge-MLP tiling: 64 edges per block, 8 waves, each wave = 4 row-tiles x 2 col-tiles
#define EPB 64

// Dynamic LDS layout for edge kernel (bytes)
#define AS_OFF  0
#define AS_BYTES   (EPB * K1 * 2)           // 102400
#define HS_OFF  (AS_OFF + AS_BYTES)         // 102400
#define HS_BYTES   (EPB * DD * 2)           // 32768
#define ES_OFF  (HS_OFF + HS_BYTES)         // 135168
#define ES_BYTES   (EPB * 17 * 4)           // 4352
#define SD_OFF  (ES_OFF + ES_BYTES)         // 139520
#define SS_OFF  (SD_OFF + EPB * 4)          // 139776
#define SMEM_TOTAL (SS_OFF + EPB * 4)       // 140032

typedef unsigned short u16;
typedef __bf16 bf16_t;
typedef bf16_t v16bf __attribute__((ext_vector_type(16)));
typedef float  v8f   __attribute__((ext_vector_type(8)));

static __device__ __forceinline__ u16 f2bfbits(float f) {
    unsigned u = __builtin_bit_cast(unsigned, f);
    return (u16)((u + 0x7FFFu + ((u >> 16) & 1u)) >> 16);   // round-to-nearest-even
}

// Build a 16-element bf16 fragment from two aligned 8-element (16B) chunks.
static __device__ __forceinline__ v16bf frag_from(const u16* p0, const u16* p1) {
    struct P { uint4 a; uint4 b; } t;
    t.a = *reinterpret_cast<const uint4*>(p0);
    t.b = *reinterpret_cast<const uint4*>(p1);
    return __builtin_bit_cast(v16bf, t);
}

static __device__ __forceinline__ v8f wmma_bf16(v16bf a, v16bf b, v8f c) {
    return __builtin_amdgcn_wmma_f32_16x16x32_bf16(false, a, false, b, (short)0, c, false, false);
}

// ---------------------------------------------------------------------------
// Kernel 0: node embedding (s = node_s @ Ws + bs, fp32 + bf16 copies)
//           + v_emb + v_norm (bf16). One node per block, 256 threads.
// ---------------------------------------------------------------------------
__global__ void node_embed_kernel(const float* __restrict__ node_s,
                                  const float* __restrict__ node_v,
                                  const float* __restrict__ Ws,
                                  const float* __restrict__ bs,
                                  const float* __restrict__ Wv,
                                  const float* __restrict__ bv,
                                  float* __restrict__ s_f32,
                                  u16*   __restrict__ s_bf,
                                  u16*   __restrict__ vn_bf) {
    __shared__ float ns[23];
    const int n = blockIdx.x;
    const int tid = threadIdx.x;
    if (tid < 23) ns[tid] = node_s[n * 23 + tid];
    __syncthreads();

    float acc = bs[tid];
    #pragma unroll
    for (int k = 0; k < 23; ++k) acc = fmaf(ns[k], Ws[k * DD + tid], acc);
    s_f32[n * DD + tid] = acc;
    s_bf [n * DD + tid] = f2bfbits(acc);

    if (tid < VDIM) {
        float nx = 0.0f;
        #pragma unroll
        for (int c = 0; c < 3; ++c) {
            float a = bv[tid];
            #pragma unroll
            for (int k = 0; k < 4; ++k)
                a = fmaf(node_v[n * 12 + k * 3 + c], Wv[k * VDIM + tid], a);
            nx = fmaf(a, a, nx);
        }
        vn_bf[n * VDIM + tid] = f2bfbits(sqrtf(nx));
    }
}

// ---------------------------------------------------------------------------
// Pack a column-weight matrix into WMMA-B-fragment-contiguous bf16 layout:
// out[((kt*16+ct)*32+lane)*16 + j] with krow = kt*32 + (lane>=16?16:0) + j,
// col = ct*16 + (lane&15).
// mode 0: Wm1 [784,256] -> K=800, zero rows inserted at 528..543
// mode 1: Wm2 [256,256] -> K=256
// mode 2: Wn  [272,256] -> K=288, rows >=272 zero
// ---------------------------------------------------------------------------
__global__ void pack_b_kernel(const float* __restrict__ W, u16* __restrict__ out,
                              int KT, int mode) {
    const int idx = blockIdx.x * blockDim.x + threadIdx.x;
    const int total = KT * 16 * 32 * 16;
    if (idx >= total) return;
    const int j    = idx & 15;
    const int lane = (idx >> 4) & 31;
    const int ct   = (idx >> 9) & 15;
    const int kt   = idx >> 13;
    const int col  = ct * 16 + (lane & 15);
    const int krow = kt * 32 + ((lane >> 4) << 4) + j;
    float val = 0.0f;
    if (mode == 0) {
        if (krow < 528)                      val = W[krow * DD + col];
        else if (krow >= 544 && krow < 800)  val = W[(krow - 16) * DD + col];
    } else if (mode == 1) {
        val = W[krow * DD + col];
    } else {
        if (krow < 272) val = W[krow * DD + col];
    }
    out[idx] = f2bfbits(val);
}

// ---------------------------------------------------------------------------
// Kernel 2: fused per-edge MLP (compute-bound core).
// 64 edges per block, 256 threads = 8 waves.
// Wave w owns col tiles {2w, 2w+1} and ALL 4 row tiles -> 8 accumulators,
// so each B fragment feeds 4 WMMAs and each A fragment feeds 2.
// Stages A = [s[dst] | s[src] | vnorm[src] | 0 | e(edge_s)] in LDS (bf16),
// GEMM1 (K=800) -> ReLU -> bf16 -> GEMM2 (K=256) -> atomic scatter into agg.
// ---------------------------------------------------------------------------
__global__ void __launch_bounds__(256)
edge_mlp_kernel(const float* __restrict__ edge_s,
                const int*   __restrict__ edge_index,
                const u16*   __restrict__ s_bf,
                const u16*   __restrict__ vn_bf,
                const float* __restrict__ We,  const float* __restrict__ be,
                const u16*   __restrict__ Wm1t, const float* __restrict__ bm1,
                const u16*   __restrict__ Wm2t, const float* __restrict__ bm2,
                float* __restrict__ agg, float* __restrict__ cnt) {
    extern __shared__ char smem[];
    u16*   As   = (u16*)  (smem + AS_OFF);   // [EPB][K1]  row stride K1 (1600B, 16B-aligned)
    u16*   Hs   = (u16*)  (smem + HS_OFF);   // [EPB][DD]
    float* Es   = (float*)(smem + ES_OFF);   // [EPB][17]
    int*   sdst = (int*)  (smem + SD_OFF);
    int*   ssrc = (int*)  (smem + SS_OFF);

    const int tid = threadIdx.x;
    const int e0  = blockIdx.x * EPB;

    if (tid < EPB) {
        ssrc[tid] = edge_index[e0 + tid];        // edge_index[0] = src
        sdst[tid] = edge_index[EE + e0 + tid];   // edge_index[1] = dst
    }
    __syncthreads();

    // gather s[dst], s[src] rows as 16B chunks (b128 traffic)
    for (int idx = tid; idx < EPB * 32; idx += 256) {
        const int r = idx >> 5, q = idx & 31;
        ((uint4*)(As + r * K1))[q] =
            ((const uint4*)(s_bf + (size_t)sdst[r] * DD))[q];
        ((uint4*)(As + r * K1 + DD))[q] =
            ((const uint4*)(s_bf + (size_t)ssrc[r] * DD))[q];
    }
    for (int idx = tid; idx < EPB * VDIM; idx += 256) {
        const int r = idx >> 4, c = idx & 15;
        As[r * K1 + 512 + c] = vn_bf[(size_t)ssrc[r] * VDIM + c];
        As[r * K1 + 528 + c] = 0;   // zero pad rows
    }
    for (int idx = tid; idx < EPB * 17; idx += 256) {
        const int r = idx / 17, k = idx % 17;
        Es[r * 17 + k] = edge_s[(size_t)(e0 + r) * 17 + k];
    }
    __syncthreads();

    // fused edge embedding: e = [edge_s,0] @ We + be  (K=17 effective)
    for (int idx = tid; idx < EPB * DD; idx += 256) {
        const int r = idx >> 8, c = idx & 255;
        float acc = be[c];
        #pragma unroll
        for (int k = 0; k < 17; ++k) acc = fmaf(Es[r * 17 + k], We[k * DD + c], acc);
        As[r * K1 + 544 + c] = f2bfbits(acc);
    }
    __syncthreads();

    const int w = tid >> 5, lane = tid & 31;
    const int h = lane >> 4, r = lane & 15;
    const int ct0 = 2 * w, ct1 = 2 * w + 1;

    // --- GEMM1: [64 x 800] @ [800 x 256] ---
    v8f acc[4][2];
    #pragma unroll
    for (int rt = 0; rt < 4; ++rt) {
        acc[rt][0] = (v8f){0,0,0,0,0,0,0,0};
        acc[rt][1] = (v8f){0,0,0,0,0,0,0,0};
    }
    for (int kt = 0; kt < KT1; ++kt) {
        const u16* bp0 = Wm1t + (((size_t)(kt * 16 + ct0) * 32 + lane) << 4);
        const u16* bp1 = Wm1t + (((size_t)(kt * 16 + ct1) * 32 + lane) << 4);
        const v16bf b0 = frag_from(bp0, bp0 + 8);
        const v16bf b1 = frag_from(bp1, bp1 + 8);
        #pragma unroll
        for (int rt = 0; rt < 4; ++rt) {
            const u16* ap = As + (rt * 16 + r) * K1 + kt * 32 + h * 16;
            const v16bf a = frag_from(ap, ap + 8);
            acc[rt][0] = wmma_bf16(a, b0, acc[rt][0]);
            acc[rt][1] = wmma_bf16(a, b1, acc[rt][1]);
        }
    }
    // bias + ReLU -> bf16 hidden tile
    #pragma unroll
    for (int rt = 0; rt < 4; ++rt) {
        #pragma unroll
        for (int i = 0; i < 8; ++i) {
            const int m  = rt * 16 + i + h * 8;
            const int n0 = ct0 * 16 + r, n1 = ct1 * 16 + r;
            Hs[m * DD + n0] = f2bfbits(fmaxf(acc[rt][0][i] + bm1[n0], 0.0f));
            Hs[m * DD + n1] = f2bfbits(fmaxf(acc[rt][1][i] + bm1[n1], 0.0f));
        }
    }
    __syncthreads();

    // --- GEMM2: [64 x 256] @ [256 x 256] ---
    #pragma unroll
    for (int rt = 0; rt < 4; ++rt) {
        acc[rt][0] = (v8f){0,0,0,0,0,0,0,0};
        acc[rt][1] = (v8f){0,0,0,0,0,0,0,0};
    }
    for (int kt = 0; kt < KT2; ++kt) {
        const u16* bp0 = Wm2t + (((size_t)(kt * 16 + ct0) * 32 + lane) << 4);
        const u16* bp1 = Wm2t + (((size_t)(kt * 16 + ct1) * 32 + lane) << 4);
        const v16bf b0 = frag_from(bp0, bp0 + 8);
        const v16bf b1 = frag_from(bp1, bp1 + 8);
        #pragma unroll
        for (int rt = 0; rt < 4; ++rt) {
            const u16* ap = Hs + (rt * 16 + r) * DD + kt * 32 + h * 16;
            const v16bf a = frag_from(ap, ap + 8);
            acc[rt][0] = wmma_bf16(a, b0, acc[rt][0]);
            acc[rt][1] = wmma_bf16(a, b1, acc[rt][1]);
        }
    }
    // scatter-add messages into agg[dst]
    #pragma unroll
    for (int rt = 0; rt < 4; ++rt) {
        #pragma unroll
        for (int i = 0; i < 8; ++i) {
            const int m  = rt * 16 + i + h * 8;
            const int dn = sdst[m];
            const int n0 = ct0 * 16 + r, n1 = ct1 * 16 + r;
            unsafeAtomicAdd(&agg[(size_t)dn * DD + n0], acc[rt][0][i] + bm2[n0]);
            unsafeAtomicAdd(&agg[(size_t)dn * DD + n1], acc[rt][1][i] + bm2[n1]);
        }
    }
    if (tid < EPB) unsafeAtomicAdd(&cnt[sdst[tid]], 1.0f);
}

// ---------------------------------------------------------------------------
// Kernel 3: residual + mean-agg, node MLP (WMMA, K=288), LayerNorm, ReLU,
// atomic global-mean-pool accumulation. 16 nodes/block, 256 threads.
// ---------------------------------------------------------------------------
__global__ void __launch_bounds__(256)
node_mlp_kernel(const float* __restrict__ s_f32,
                const float* __restrict__ agg, const float* __restrict__ cnt,
                const u16*   __restrict__ vn_bf,
                const int*   __restrict__ batch,
                const u16*   __restrict__ Wnt, const float* __restrict__ bn,
                const float* __restrict__ gamma, const float* __restrict__ beta,
                float* __restrict__ gsum, float* __restrict__ gcnt) {
    __shared__ u16   Fs[16][K3];
    __shared__ float Hf[16][DD];
    __shared__ float invc[16], mu[16], rs[16];
    __shared__ int   nb[16];

    const int tid = threadIdx.x;
    const int n0  = blockIdx.x * 16;

    if (tid < 16) {
        const int node = n0 + tid;
        invc[tid] = 1.0f / fmaxf(cnt[node], 1.0f);
        nb[tid]   = batch[node];
    }
    __syncthreads();

    for (int idx = tid; idx < 16 * DD; idx += 256) {
        const int r = idx >> 8, c = idx & 255;
        const size_t o = (size_t)(n0 + r) * DD + c;
        Fs[r][c] = f2bfbits(s_f32[o] + agg[o] * invc[r]);   // residual + mean aggr
    }
    {
        const int r = tid >> 4, c = tid & 15;
        Fs[r][DD + c]      = vn_bf[(size_t)(n0 + r) * VDIM + c];
        Fs[r][DD + 16 + c] = 0;
    }
    __syncthreads();

    const int w = tid >> 5, lane = tid & 31;
    const int h = lane >> 4, r = lane & 15;
    const int ct0 = 2 * w, ct1 = 2 * w + 1;

    v8f c0 = {0,0,0,0,0,0,0,0}, c1 = {0,0,0,0,0,0,0,0};
    for (int kt = 0; kt < KT3; ++kt) {
        const u16* ap = &Fs[r][kt * 32 + h * 16];
        v16bf a = frag_from(ap, ap + 8);
        const u16* bp0 = Wnt + (((size_t)(kt * 16 + ct0) * 32 + lane) << 4);
        const u16* bp1 = Wnt + (((size_t)(kt * 16 + ct1) * 32 + lane) << 4);
        c0 = wmma_bf16(a, frag_from(bp0, bp0 + 8), c0);
        c1 = wmma_bf16(a, frag_from(bp1, bp1 + 8), c1);
    }
    #pragma unroll
    for (int i = 0; i < 8; ++i) {
        const int m = i + h * 8;
        const int n0c = ct0 * 16 + r, n1c = ct1 * 16 + r;
        Hf[m][n0c] = c0[i] + bn[n0c];
        Hf[m][n1c] = c1[i] + bn[n1c];
    }
    __syncthreads();

    if (tid < 16) {
        float s = 0.0f, sq = 0.0f;
        for (int c = 0; c < DD; ++c) { float v = Hf[tid][c]; s += v; sq = fmaf(v, v, sq); }
        const float m = s * (1.0f / DD);
        mu[tid] = m;
        rs[tid] = rsqrtf(sq * (1.0f / DD) - m * m + 1e-5f);
    }
    __syncthreads();

    for (int idx = tid; idx < 16 * DD; idx += 256) {
        const int r2 = idx >> 8, c = idx & 255;
        float v = (Hf[r2][c] - mu[r2]) * rs[r2] * gamma[c] + beta[c];
        v = fmaxf(v, 0.0f);
        unsafeAtomicAdd(&gsum[(size_t)nb[r2] * DD + c], v);
    }
    if (tid < 16) unsafeAtomicAdd(&gcnt[nb[tid]], 1.0f);
}

// ---------------------------------------------------------------------------
// Kernel 4: final pool division
// ---------------------------------------------------------------------------
__global__ void pool_div_kernel(const float* __restrict__ gsum,
                                const float* __restrict__ gcnt,
                                float* __restrict__ out) {
    const int idx = blockIdx.x * blockDim.x + threadIdx.x;
    if (idx < GG * DD) out[idx] = gsum[idx] / fmaxf(gcnt[idx >> 8], 1.0f);
}

// ---------------------------------------------------------------------------
extern "C" void kernel_launch(void* const* d_in, const int* in_sizes, int n_in,
                              void* d_out, int out_size, void* d_ws, size_t ws_size,
                              hipStream_t stream) {
    const float* node_s     = (const float*)d_in[0];
    const float* node_v     = (const float*)d_in[1];
    const float* edge_s     = (const float*)d_in[2];
    const int*   edge_index = (const int*)  d_in[3];
    const int*   batch      = (const int*)  d_in[4];
    const float* Ws  = (const float*)d_in[5];
    const float* bs  = (const float*)d_in[6];
    const float* Wv  = (const float*)d_in[7];
    const float* bv  = (const float*)d_in[8];
    const float* We  = (const float*)d_in[9];
    const float* be  = (const float*)d_in[10];
    const float* Wm1 = (const float*)d_in[11];
    const float* bm1 = (const float*)d_in[12];
    const float* Wm2 = (const float*)d_in[13];
    const float* bm2 = (const float*)d_in[14];
    const float* Wn  = (const float*)d_in[15];
    const float* bn  = (const float*)d_in[16];
    const float* gamma = (const float*)d_in[17];
    const float* beta  = (const float*)d_in[18];
    float* out = (float*)d_out;

    // ---- workspace carve (256B aligned regions) ----
    char* base = (char*)d_ws;
    size_t off = 0;
    auto carve = [&](size_t bytes) -> void* {
        off = (off + 255) & ~(size_t)255;
        void* p = base + off;
        off += bytes;
        return p;
    };
    float* s_f32 = (float*)carve((size_t)NN * DD * 4);
    u16*   s_bf  = (u16*)  carve((size_t)NN * DD * 2);
    u16*   vn_bf = (u16*)  carve((size_t)NN * VDIM * 2);
    float* agg   = (float*)carve((size_t)NN * DD * 4);
    float* cnt   = (float*)carve((size_t)NN * 4);
    float* gsum  = (float*)carve((size_t)GG * DD * 4);
    float* gcnt  = (float*)carve((size_t)GG * 4);
    u16*   Wm1t  = (u16*)  carve((size_t)KT1 * 16 * 32 * 16 * 2);
    u16*   Wm2t  = (u16*)  carve((size_t)KT2 * 16 * 32 * 16 * 2);
    u16*   Wnt   = (u16*)  carve((size_t)KT3 * 16 * 32 * 16 * 2);

    // ---- zero accumulators ----
    hipMemsetAsync(agg,  0, (size_t)NN * DD * 4, stream);
    hipMemsetAsync(cnt,  0, (size_t)NN * 4,      stream);
    hipMemsetAsync(gsum, 0, (size_t)GG * DD * 4, stream);
    hipMemsetAsync(gcnt, 0, (size_t)GG * 4,      stream);

    // ---- weight pre-swizzle into WMMA B-fragment layout ----
    {
        int t1 = KT1 * 16 * 32 * 16;
        int t2 = KT2 * 16 * 32 * 16;
        int t3 = KT3 * 16 * 32 * 16;
        pack_b_kernel<<<(t1 + 255) / 256, 256, 0, stream>>>(Wm1, Wm1t, KT1, 0);
        pack_b_kernel<<<(t2 + 255) / 256, 256, 0, stream>>>(Wm2, Wm2t, KT2, 1);
        pack_b_kernel<<<(t3 + 255) / 256, 256, 0, stream>>>(Wn,  Wnt,  KT3, 2);
    }

    // ---- pipeline ----
    node_embed_kernel<<<NN, 256, 0, stream>>>(node_s, node_v, Ws, bs, Wv, bv,
                                              s_f32, s_bf, vn_bf);
    edge_mlp_kernel<<<EE / EPB, 256, SMEM_TOTAL, stream>>>(edge_s, edge_index,
                                                 s_bf, vn_bf,
                                                 We, be, Wm1t, bm1, Wm2t, bm2,
                                                 agg, cnt);
    node_mlp_kernel<<<NN / 16, 256, 0, stream>>>(s_f32, agg, cnt, vn_bf, batch,
                                                 Wnt, bn, gamma, beta, gsum, gcnt);
    pool_div_kernel<<<GG, 256, 0, stream>>>(gsum, gcnt, out);
}
